// EEGConnectomeGNN_69750268887652
// MI455X (gfx1250) — matrix operand order repaired and checked by reading, not compiled, not run on hardware
//
#include <hip/hip_runtime.h>
#include <hip/hip_bf16.h>

typedef __attribute__((ext_vector_type(16))) _Float16 v16h;
typedef __attribute__((ext_vector_type(8)))  _Float16 v8h;
typedef __attribute__((ext_vector_type(8)))  float    v8f;

union F16x16 { v16h v; v8h h[2]; };

#define NNODES 40000
#define NEDGES 640000
#define DIM    128
#define NGRAPH 64
#define LDA    136   // padded LDS row stride (f16 elems): 272B = 17*16B, keeps 16B alignment

// ---------------------------------------------------------------- utilities
__global__ __launch_bounds__(256) void zero_f32_kernel(float* __restrict__ p, int n) {
    int i = blockIdx.x * 256 + threadIdx.x;
    if (i < n) p[i] = 0.0f;
}

// Convert fp32 weight W[k][n] -> f16 TRANSPOSED Wt[n][k] (so B fragments are
// two runs of 8 contiguous K-values at fixed N).
__global__ __launch_bounds__(256) void cvt_w_transpose_kernel(const float* __restrict__ src,
                                                              _Float16* __restrict__ dst) {
    int i = blockIdx.x * 256 + threadIdx.x;   // i = n*128 + k
    if (i < DIM * DIM) {
        int n = i >> 7, k = i & 127;
        dst[i] = (_Float16)src[k * DIM + n];
    }
}

// ------------------------------------------------- message + scatter (GINE)
// One wave (32 lanes) per edge; each lane handles 4 contiguous floats.
// msg = relu(h[src] + edge_attr[e]); atomic scatter-add into agg[dst].
__global__ __launch_bounds__(256) void scatter_msg_kernel(const float* __restrict__ hin,
                                                          const float* __restrict__ ea,
                                                          const int*   __restrict__ ei,
                                                          float*       __restrict__ agg,
                                                          int nedges) {
    int e = blockIdx.x * 8 + (threadIdx.x >> 5);
    if (e >= nedges) return;
    int lane = threadIdx.x & 31;
    int src = ei[e];
    int dst = ei[nedges + e];
    int c = lane * 4;
    // prefetch the streaming edge_attr a few KB ahead (global_prefetch_b8)
    if (e + 64 < nedges)
        __builtin_prefetch(ea + (size_t)(e + 64) * DIM + c, 0, 1);
    const float4 xv = *(const float4*)(hin + (size_t)src * DIM + c);
    const float4 av = *(const float4*)(ea  + (size_t)e   * DIM + c);
    float m0 = xv.x + av.x; m0 = m0 > 0.f ? m0 : 0.f;
    float m1 = xv.y + av.y; m1 = m1 > 0.f ? m1 : 0.f;
    float m2 = xv.z + av.z; m2 = m2 > 0.f ? m2 : 0.f;
    float m3 = xv.w + av.w; m3 = m3 > 0.f ? m3 : 0.f;
    float* ap = agg + (size_t)dst * DIM + c;
    atomicAdd(ap + 0, m0);
    atomicAdd(ap + 1, m1);
    atomicAdd(ap + 2, m2);
    atomicAdd(ap + 3, m3);
}

// -------------------------------------------- fused 2-GEMM MLP via WMMA f16
// Block: 256 threads = 8 waves, 128 node-rows per block, each wave owns 16 rows.
// hout = relu( relu( (hin+agg) @ W1 + b1 ) @ W2 + b2 )
__global__ __launch_bounds__(256) void gine_mlp_wmma_kernel(const float*    __restrict__ hin,
                                                            const float*    __restrict__ agg,
                                                            const _Float16* __restrict__ w1t,
                                                            const float*    __restrict__ b1,
                                                            const _Float16* __restrict__ w2t,
                                                            const float*    __restrict__ b2,
                                                            float*          __restrict__ hout,
                                                            int nrows) {
    __shared__ _Float16 sA[DIM * LDA];   // activations, row-major [row][k]
    __shared__ _Float16 sWt[DIM * LDA];  // weights, transposed [n][k]; W1 then W2
    __shared__ _Float16 sT[DIM * LDA];   // hidden activations, row-major [row][k]

    const int tid  = threadIdx.x;
    const int wave = tid >> 5;      // 0..7
    const int lane = tid & 31;
    const int half = lane >> 4;     // 0/1
    const int l16  = lane & 15;
    const int row0 = blockIdx.x * 128;

    // Stage A: (hin + agg) -> sA as f16. float4 global loads, 64b LDS stores.
    for (int i = tid; i < DIM * DIM / 4; i += 256) {
        int r = i >> 5, c = (i & 31) * 4;
        int gr = row0 + r;
        float4 xv = make_float4(0.f, 0.f, 0.f, 0.f);
        if (gr < nrows) {
            float4 hv = *(const float4*)(hin + (size_t)gr * DIM + c);
            float4 av = *(const float4*)(agg + (size_t)gr * DIM + c);
            xv = make_float4(hv.x + av.x, hv.y + av.y, hv.z + av.z, hv.w + av.w);
        }
        _Float16* p = sA + r * LDA + c;
        p[0] = (_Float16)xv.x; p[1] = (_Float16)xv.y;
        p[2] = (_Float16)xv.z; p[3] = (_Float16)xv.w;
    }
    // Stage W1t -> sWt (contiguous v8h copies)
    for (int i = tid; i < DIM * DIM / 8; i += 256) {
        int r = (i * 8) >> 7, c = (i * 8) & 127;
        *(v8h*)(sWt + r * LDA + c) = *(const v8h*)(w1t + i * 8);
    }
    __syncthreads();

    const int arow = wave * 16 + l16;   // A-matrix row for this lane (both halves)

    // ---------------- GEMM1: T = relu(A @ W1 + b1) ----------------
    {
        // Hoist K-step A fragments into registers (nt-invariant).
        F16x16 a[4];
#pragma unroll
        for (int kt = 0; kt < 4; ++kt) {
            const int k0 = kt * 32;
            a[kt].h[0] = *(const v8h*)(sA + arow * LDA + k0 + half * 8);
            a[kt].h[1] = *(const v8h*)(sA + arow * LDA + k0 + 16 + half * 8);
        }
        for (int nt = 0; nt < 8; ++nt) {
            const int n0 = nt * 16;
            const _Float16* brow = sWt + (n0 + l16) * LDA + half * 8;
            v8f c = {};
#pragma unroll
            for (int kt = 0; kt < 4; ++kt) {
                const int k0 = kt * 32;
                F16x16 b;
                b.h[0] = *(const v8h*)(brow + k0);
                b.h[1] = *(const v8h*)(brow + k0 + 16);
                c = __builtin_amdgcn_wmma_f32_16x16x32_f16(false, a[kt].v, false, b.v,
                                                           (short)0, c, false, false);
            }
            const float bias = b1[n0 + l16];
#pragma unroll
            for (int g = 0; g < 8; ++g) {
                int r = wave * 16 + half * 8 + g;   // C/D row layout
                float v = c[g] + bias;
                v = v > 0.f ? v : 0.f;
                sT[r * LDA + n0 + l16] = (_Float16)v;
            }
        }
    }
    __syncthreads();           // all waves done reading sWt

    // Stage W2t -> sWt
    for (int i = tid; i < DIM * DIM / 8; i += 256) {
        int r = (i * 8) >> 7, c = (i * 8) & 127;
        *(v8h*)(sWt + r * LDA + c) = *(const v8h*)(w2t + i * 8);
    }
    __syncthreads();

    // ---------------- GEMM2: out = relu(T @ W2 + b2) ----------------
    {
        F16x16 a[4];
#pragma unroll
        for (int kt = 0; kt < 4; ++kt) {
            const int k0 = kt * 32;
            a[kt].h[0] = *(const v8h*)(sT + arow * LDA + k0 + half * 8);
            a[kt].h[1] = *(const v8h*)(sT + arow * LDA + k0 + 16 + half * 8);
        }
        for (int nt = 0; nt < 8; ++nt) {
            const int n0 = nt * 16;
            const _Float16* brow = sWt + (n0 + l16) * LDA + half * 8;
            v8f c = {};
#pragma unroll
            for (int kt = 0; kt < 4; ++kt) {
                const int k0 = kt * 32;
                F16x16 b;
                b.h[0] = *(const v8h*)(brow + k0);
                b.h[1] = *(const v8h*)(brow + k0 + 16);
                c = __builtin_amdgcn_wmma_f32_16x16x32_f16(false, a[kt].v, false, b.v,
                                                           (short)0, c, false, false);
            }
            const float bias = b2[n0 + l16];
#pragma unroll
            for (int g = 0; g < 8; ++g) {
                int gr = row0 + wave * 16 + half * 8 + g;
                if (gr < nrows) {
                    float v = c[g] + bias;
                    v = v > 0.f ? v : 0.f;
                    hout[(size_t)gr * DIM + n0 + l16] = v;
                }
            }
        }
    }
}

// ------------------------------------------------------- global mean pool
__global__ __launch_bounds__(128) void pool_acc_kernel(const float* __restrict__ h,
                                                       const int*   __restrict__ batch,
                                                       float* __restrict__ sums,
                                                       float* __restrict__ cnt,
                                                       int nrows) {
    int node = blockIdx.x;
    if (node >= nrows) return;
    int c = threadIdx.x;
    int g = batch[node];
    atomicAdd(&sums[g * DIM + c], h[(size_t)node * DIM + c]);
    if (c == 0) atomicAdd(&cnt[g], 1.0f);
}

__global__ __launch_bounds__(128) void pool_fin_kernel(const float* __restrict__ sums,
                                                       const float* __restrict__ cnt,
                                                       float* __restrict__ out) {
    int i = blockIdx.x * 128 + threadIdx.x;   // 64*128
    int g = i >> 7;
    float cc = cnt[g];
    cc = cc < 1.0f ? 1.0f : cc;
    out[i] = sums[i] / cc;
}

// ------------------------------------------------------------------ driver
extern "C" void kernel_launch(void* const* d_in, const int* in_sizes, int n_in,
                              void* d_out, int out_size, void* d_ws, size_t ws_size,
                              hipStream_t stream) {
    (void)in_sizes; (void)n_in; (void)out_size; (void)ws_size;

    const float* x     = (const float*)d_in[0];
    const float* ea    = (const float*)d_in[1];
    const int*   ei    = (const int*)d_in[2];
    const int*   batch = (const int*)d_in[3];
    // per-layer params: d_in[4 + 4*l + {0:w1,1:b1,2:w2,3:b2}]

    char* ws = (char*)d_ws;
    const size_t HBYTES = (size_t)NNODES * DIM * sizeof(float);   // 20.48 MB
    float*    agg  = (float*)(ws);
    float*    hA   = (float*)(ws + HBYTES);
    float*    hB   = (float*)(ws + 2 * HBYTES);
    _Float16* w16  = (_Float16*)(ws + 3 * HBYTES);                // 8 * 16384 f16 (transposed)
    float*    sums = (float*)(ws + 3 * HBYTES + 8 * DIM * DIM * sizeof(_Float16));
    float*    cnt  = sums + NGRAPH * DIM;

    // Convert all 8 weight matrices fp32 [k][n] -> f16 transposed [n][k]
    for (int l = 0; l < 4; ++l) {
        cvt_w_transpose_kernel<<<(DIM * DIM + 255) / 256, 256, 0, stream>>>(
            (const float*)d_in[4 + 4 * l + 0], w16 + (2 * l + 0) * DIM * DIM);
        cvt_w_transpose_kernel<<<(DIM * DIM + 255) / 256, 256, 0, stream>>>(
            (const float*)d_in[4 + 4 * l + 2], w16 + (2 * l + 1) * DIM * DIM);
    }

    const float* hin = x;
    float* hout = hA;
    const int aggN = NNODES * DIM;
    for (int l = 0; l < 4; ++l) {
        zero_f32_kernel<<<(aggN + 255) / 256, 256, 0, stream>>>(agg, aggN);
        scatter_msg_kernel<<<(NEDGES + 7) / 8, 256, 0, stream>>>(hin, ea, ei, agg, NEDGES);
        gine_mlp_wmma_kernel<<<(NNODES + 127) / 128, 256, 0, stream>>>(
            hin, agg,
            w16 + (2 * l + 0) * DIM * DIM, (const float*)d_in[4 + 4 * l + 1],
            w16 + (2 * l + 1) * DIM * DIM, (const float*)d_in[4 + 4 * l + 3],
            hout, NNODES);
        hin  = hout;
        hout = (hout == hA) ? hB : hA;
    }

    // mean pool
    zero_f32_kernel<<<(NGRAPH * DIM + NGRAPH + 255) / 256, 256, 0, stream>>>(
        sums, NGRAPH * DIM + NGRAPH);
    pool_acc_kernel<<<NNODES, 128, 0, stream>>>(hin, batch, sums, cnt, NNODES);
    pool_fin_kernel<<<NGRAPH, 128, 0, stream>>>(sums, cnt, (float*)d_out);
}
